// VPRGraphEncoder_12893491823321
// MI455X (gfx1250) — matrix operand order repaired and checked by reading, not compiled, not run on hardware
//
#include <hip/hip_runtime.h>
#include <hip/hip_bf16.h>
#include <math.h>

typedef __attribute__((ext_vector_type(16))) __bf16 v16bf;
typedef __attribute__((ext_vector_type(8)))  float  v8f;

#define HDIM 128          // hidden dim (output cols of every node GEMM)
#define TILE_ROWS 128     // rows per block (8 waves x 16 rows)
#define KSTAGE 160        // max K columns staged in LDS per pass (320 = 2 passes, 128 = 1)

// ---------------------------------------------------------------------------
// Utility kernels
// ---------------------------------------------------------------------------
__global__ void fill_f32(float* __restrict__ p, float v, long long n) {
    long long i = (long long)blockIdx.x * blockDim.x + threadIdx.x;
    if (i < n) p[i] = v;
}

__global__ void deg_count_kernel(const int* __restrict__ cols, float* __restrict__ deg, int E) {
    int e = blockIdx.x * blockDim.x + threadIdx.x;
    if (e < E) atomicAdd(&deg[cols[e]], 1.0f);
}

__global__ void dinv_kernel(float* __restrict__ deg, int N) {
    int i = blockIdx.x * blockDim.x + threadIdx.x;
    if (i < N) deg[i] = rsqrtf(deg[i]);   // deg >= 1 always (self loop)
}

// ---------------------------------------------------------------------------
// WMMA GEMM over nodes: Out[N x 128] = act( A_eff[N x K] @ W[K x 128] + bias )
// A_eff is either a plain matrix A, or (concat=1) the concat of X[N x 256]
// with emb[cls[n]][0..63]  (K = 320).
// One block = 256 threads = 8 waves; wave w computes rows m0..m0+15, all 128
// output columns as 8 independent 16x16 WMMA accumulators.
// W^T is staged in LDS as bf16 so B-fragments are 16 contiguous elements.
// Out-of-range rows are clamped (computed but never stored) so the inner loop
// has no per-lane predication: A loads are plain float4 (global_load_b128).
// ---------------------------------------------------------------------------
__global__ __launch_bounds__(256)
void gemm_node_kernel(const float* __restrict__ A,
                      const float* __restrict__ X,
                      const int*   __restrict__ cls,
                      const float* __restrict__ emb,
                      const float* __restrict__ W,      // [K x 128] row-major
                      const float* __restrict__ bias,   // [128] or nullptr
                      float* __restrict__ Out,          // [N x 128]
                      int N, int K, int doRelu, int concat)
{
    __shared__ __bf16 wT[HDIM * KSTAGE];   // wT[n*KSTAGE + klocal] = W[k][n], 40 KB

    const int tid   = threadIdx.x;
    const int wave  = tid >> 5;
    const int lane  = tid & 31;
    const int lrow  = lane & 15;     // 0..15
    const int lhalf = lane >> 4;     // 0 or 1
    const int m0    = blockIdx.x * TILE_ROWS + wave * 16;
    int arow = m0 + lrow;
    if (arow >= N) arow = N - 1;     // clamp: garbage rows computed, never stored

    // Hoisted per-row source pointers (cls gather happens exactly once).
    const float* rowA = concat ? nullptr : (A + (long long)arow * K);
    const float* rowX = concat ? (X + (long long)arow * 256) : nullptr;
    const float* rowE = concat ? (emb + (long long)cls[arow] * 64) : nullptr;

    v8f acc[8] = {};

    for (int kbase = 0; kbase < K; kbase += KSTAGE) {
        const int klen = min(KSTAGE, K - kbase);     // 160 or 128 (multiple of 32)

        // Stage W^T (bf16).  Consecutive threads read consecutive n -> coalesced.
        for (int idx = tid; idx < HDIM * klen; idx += 256) {
            int n  = idx & (HDIM - 1);
            int kl = idx >> 7;
            wT[n * KSTAGE + kl] = (__bf16)W[(long long)(kbase + kl) * HDIM + n];
        }
        __syncthreads();

        for (int kc = 0; kc < klen; kc += 32) {
            // Chunk [kglob, kglob+32) lies entirely in one source (32-aligned
            // boundary at 256), so the source select is one uniform branch.
            const int kglob = kbase + kc;
            const float* src;
            if (!concat)          src = rowA + kglob;
            else if (kglob < 256) src = rowX + kglob;
            else                  src = rowE + (kglob - 256);

            // ---- A fragment: contiguous k order, k = 16*lhalf + s (B matches) ----
            // lanes 0-15  : row M=lane,    K = kc..kc+15
            // lanes 16-31 : row M=lane-16, K = kc+16..kc+31
            const float4 fa = *(const float4*)(src + lhalf * 16);
            const float4 fb = *(const float4*)(src + lhalf * 16 + 4);
            const float4 fc = *(const float4*)(src + lhalf * 16 + 8);
            const float4 fd = *(const float4*)(src + lhalf * 16 + 12);
            v16bf afrag;
            afrag[0]  = (__bf16)fa.x; afrag[1]  = (__bf16)fa.y;
            afrag[2]  = (__bf16)fa.z; afrag[3]  = (__bf16)fa.w;
            afrag[4]  = (__bf16)fb.x; afrag[5]  = (__bf16)fb.y;
            afrag[6]  = (__bf16)fb.z; afrag[7]  = (__bf16)fb.w;
            afrag[8]  = (__bf16)fc.x; afrag[9]  = (__bf16)fc.y;
            afrag[10] = (__bf16)fc.z; afrag[11] = (__bf16)fc.w;
            afrag[12] = (__bf16)fd.x; afrag[13] = (__bf16)fd.y;
            afrag[14] = (__bf16)fd.z; afrag[15] = (__bf16)fd.w;

            // ---- load ALL 8 B fragments first, then burst the 8 WMMAs ----
            const int kloc = kc + lhalf * 16;   // lanes 0-15: K=kc..+15; 16-31: +16
            v16bf bfr[8];
            #pragma unroll
            for (int t = 0; t < 8; ++t) {
                const __bf16* bp = &wT[(t * 16 + lrow) * KSTAGE + kloc];
                #pragma unroll
                for (int i = 0; i < 16; ++i) bfr[t][i] = bp[i];
            }
            #pragma unroll
            for (int t = 0; t < 8; ++t)
                acc[t] = __builtin_amdgcn_wmma_f32_16x16x32_bf16(
                            false, afrag, false, bfr[t],
                            (short)0, acc[t], false, false);
        }
        __syncthreads();
    }

    // ---- epilogue: C layout lanes 0-15: M=r, lanes 16-31: M=8+r; N=lane&15 ----
    #pragma unroll
    for (int t = 0; t < 8; ++t) {
        const int col = t * 16 + lrow;
        const float b = bias ? bias[col] : 0.f;
        #pragma unroll
        for (int r = 0; r < 8; ++r) {
            const int row = m0 + r + lhalf * 8;
            if (row < N) {
                float v = acc[t][r] + b;
                if (doRelu) v = fmaxf(v, 0.f);
                Out[(long long)row * HDIM + col] = v;
            }
        }
    }
}

// ---------------------------------------------------------------------------
// Edge scatter: acc[col] += m[row] * dinv[row]*dinv[col]   (128 feature lanes)
// Edge records staged in LDS once per block instead of 128 redundant loads.
// ---------------------------------------------------------------------------
#define EDGES_PER_BLOCK 32
__global__ __launch_bounds__(128)
void scatter_kernel(const float* __restrict__ m, const float* __restrict__ dinv,
                    const int* __restrict__ rows, const int* __restrict__ cols,
                    float* __restrict__ acc, int E)
{
    __shared__ int   sr[EDGES_PER_BLOCK];
    __shared__ int   sc[EDGES_PER_BLOCK];
    __shared__ float sn[EDGES_PER_BLOCK];
    const int j  = threadIdx.x;                  // feature index 0..127
    const int e0 = blockIdx.x * EDGES_PER_BLOCK;
    if (j < EDGES_PER_BLOCK) {
        const int e = e0 + j;
        if (e < E) {
            const int r = rows[e], c = cols[e];
            sr[j] = r; sc[j] = c;
            sn[j] = dinv[r] * dinv[c];
        }
    }
    __syncthreads();
    const int cnt = min(EDGES_PER_BLOCK, E - e0);
    for (int t = 0; t < cnt; ++t) {
        atomicAdd(&acc[(long long)sc[t] * HDIM + j],
                  m[(long long)sr[t] * HDIM + j] * sn[t]);
    }
}

// h = relu(acc + m * dinv^2 (self loop) + bias)
__global__ void selfloop_bias_relu_kernel(const float* __restrict__ acc,
                                          const float* __restrict__ m,
                                          const float* __restrict__ dinv,
                                          const float* __restrict__ bias,
                                          float* __restrict__ h, int N)
{
    long long idx = (long long)blockIdx.x * blockDim.x + threadIdx.x;
    if (idx < (long long)N * HDIM) {
        int n = (int)(idx >> 7);
        int j = (int)(idx & (HDIM - 1));
        float d = dinv[n];
        float v = acc[idx] + m[idx] * d * d + bias[j];
        h[idx] = fmaxf(v, 0.f);
    }
}

// ---------------------------------------------------------------------------
// Pooling: per-graph sum, count, and max (uint bit trick: values >= 0)
// ---------------------------------------------------------------------------
#define NODES_PER_BLOCK 32
__global__ __launch_bounds__(128)
void pool_kernel(const float* __restrict__ h, const int* __restrict__ batch,
                 float* __restrict__ gsum, float* __restrict__ gcnt,
                 unsigned* __restrict__ gmax, int N)
{
    const int j  = threadIdx.x;
    const int n0 = blockIdx.x * NODES_PER_BLOCK;
    for (int t = 0; t < NODES_PER_BLOCK; ++t) {
        const int n = n0 + t;
        if (n >= N) break;
        const int g = batch[n];
        const float v = h[(long long)n * HDIM + j];
        atomicAdd(&gsum[(long long)g * HDIM + j], v);
        atomicMax(&gmax[(long long)g * HDIM + j], __float_as_uint(v));
        if (j == 0) atomicAdd(&gcnt[g], 1.0f);
    }
}

// z1[g] = relu( concat(mean, max) @ proj_w1 + b1 )       one block per graph
__global__ __launch_bounds__(128)
void proj1_kernel(const float* __restrict__ gsum, const float* __restrict__ gcnt,
                  const unsigned* __restrict__ gmax,
                  const float* __restrict__ w1, const float* __restrict__ b1,
                  float* __restrict__ z1)
{
    __shared__ float hg[2 * HDIM];
    const int g = blockIdx.x, j = threadIdx.x;
    const float c = fmaxf(gcnt[g], 1.0f);
    hg[j]        = gsum[(long long)g * HDIM + j] / c;
    hg[HDIM + j] = __uint_as_float(gmax[(long long)g * HDIM + j]);
    __syncthreads();
    float a = b1[j];
    for (int k = 0; k < 2 * HDIM; ++k) a += hg[k] * w1[k * HDIM + j];
    z1[(long long)g * HDIM + j] = fmaxf(a, 0.f);
}

// z = (z1 @ proj_w2 + b2) / max(||.||, 1e-12)            one block per graph
__global__ __launch_bounds__(128)
void proj2_norm_kernel(const float* __restrict__ z1, const float* __restrict__ w2,
                       const float* __restrict__ b2, float* __restrict__ out)
{
    __shared__ float zs[HDIM];
    __shared__ float red[HDIM];
    const int g = blockIdx.x, j = threadIdx.x;
    zs[j] = z1[(long long)g * HDIM + j];
    __syncthreads();
    float a = b2[j];
    for (int k = 0; k < HDIM; ++k) a += zs[k] * w2[k * HDIM + j];
    red[j] = a * a;
    __syncthreads();
    for (int s = 64; s > 0; s >>= 1) {
        if (j < s) red[j] += red[j + s];
        __syncthreads();
    }
    const float nrm = sqrtf(red[0]);
    out[(long long)g * HDIM + j] = a / fmaxf(nrm, 1e-12f);
}

// ---------------------------------------------------------------------------
extern "C" void kernel_launch(void* const* d_in, const int* in_sizes, int n_in,
                              void* d_out, int out_size, void* d_ws, size_t ws_size,
                              hipStream_t stream)
{
    const float* x      = (const float*)d_in[0];
    const int*   cls    = (const int*)  d_in[1];
    const int*   eidx   = (const int*)  d_in[2];
    const int*   batch  = (const int*)  d_in[3];
    const float* emb    = (const float*)d_in[4];
    const float* in_w1  = (const float*)d_in[5];
    const float* in_b1  = (const float*)d_in[6];
    const float* in_w2  = (const float*)d_in[7];
    const float* in_b2  = (const float*)d_in[8];
    const float* conv_w = (const float*)d_in[9];
    const float* conv_b = (const float*)d_in[10];
    const float* pw1    = (const float*)d_in[11];
    const float* pb1    = (const float*)d_in[12];
    const float* pw2    = (const float*)d_in[13];
    const float* pb2    = (const float*)d_in[14];
    float* out = (float*)d_out;

    const int N = in_sizes[0] / 256;
    const int E = in_sizes[2] / 2;
    const int L = in_sizes[9] / (HDIM * HDIM);
    const int G = out_size / HDIM;

    const int* rowsE = eidx;        // edge_index[0] (sources)
    const int* colsE = eidx + E;    // edge_index[1] (targets)

    // workspace carving
    float* buf0 = (float*)d_ws;                       // m            [N x 128]
    float* buf1 = buf0 + (long long)N * HDIM;         // h            [N x 128]
    float* buf2 = buf1 + (long long)N * HDIM;         // scatter acc  [N x 128]
    float* dinv = buf2 + (long long)N * HDIM;         // [N]
    float* gsum = dinv + N;                           // [G x 128]
    float* gcnt = gsum + (long long)G * HDIM;         // [G]
    unsigned* gmax = (unsigned*)(gcnt + G);           // [G x 128]
    float* z1   = (float*)(gmax + (long long)G * HDIM); // [G x 128]

    const long long NH = (long long)N * HDIM;
    const int gemmBlocks = (N + TILE_ROWS - 1) / TILE_ROWS;

    // 1) degrees with self loops -> dinv
    fill_f32<<<(N + 255) / 256, 256, 0, stream>>>(dinv, 1.0f, N);
    deg_count_kernel<<<(E + 255) / 256, 256, 0, stream>>>(colsE, dinv, E);
    dinv_kernel<<<(N + 255) / 256, 256, 0, stream>>>(dinv, N);

    // 2) input MLP: concat GEMM (K=320) then K=128 GEMM, both WMMA
    gemm_node_kernel<<<gemmBlocks, 256, 0, stream>>>(
        nullptr, x, cls, emb, in_w1, in_b1, buf0, N, 320, 1, 1);
    gemm_node_kernel<<<gemmBlocks, 256, 0, stream>>>(
        buf0, nullptr, nullptr, nullptr, in_w2, in_b2, buf1, N, HDIM, 1, 0);

    // 3) GCN layers: h(buf1) -> m(buf0) -> acc(buf2) -> h(buf1)
    for (int i = 0; i < L; ++i) {
        gemm_node_kernel<<<gemmBlocks, 256, 0, stream>>>(
            buf1, nullptr, nullptr, nullptr,
            conv_w + (long long)i * HDIM * HDIM, nullptr, buf0, N, HDIM, 0, 0);
        fill_f32<<<(int)((NH + 255) / 256), 256, 0, stream>>>(buf2, 0.0f, NH);
        scatter_kernel<<<(E + EDGES_PER_BLOCK - 1) / EDGES_PER_BLOCK, 128, 0, stream>>>(
            buf0, dinv, rowsE, colsE, buf2, E);
        selfloop_bias_relu_kernel<<<(int)((NH + 255) / 256), 256, 0, stream>>>(
            buf2, buf0, dinv, conv_b + (long long)i * HDIM, buf1, N);
    }

    // 4) pooling (sum + count + max share one contiguous zeroed region)
    fill_f32<<<(G * HDIM * 2 + G + 255) / 256, 256, 0, stream>>>(
        gsum, 0.0f, (long long)G * HDIM * 2 + G);
    pool_kernel<<<(N + NODES_PER_BLOCK - 1) / NODES_PER_BLOCK, 128, 0, stream>>>(
        buf1, batch, gsum, gcnt, gmax, N);

    // 5) projection head + L2 normalize
    proj1_kernel<<<G, 128, 0, stream>>>(gsum, gcnt, gmax, pw1, pb1, z1);
    proj2_norm_kernel<<<G, 128, 0, stream>>>(z1, pw2, pb2, out);
}
